// MultiScaleRetention_35278861369587
// MI455X (gfx1250) — compile-verified, matrix operand order checked by reading, not run on hardware
//
#include <hip/hip_runtime.h>
#include <hip/hip_bf16.h>
#include <math.h>

// ---------------------------------------------------------------------------
// MultiScaleRetention for MI455X (gfx1250, wave32, WMMA + TDM double-buffer)
//   D_MODEL=1024, N_HEADS=16, D_HEAD=64, B=2, S=2048, DECAY=0.99
// ---------------------------------------------------------------------------

typedef __attribute__((ext_vector_type(16))) __bf16 v16bf;
typedef __attribute__((ext_vector_type(8)))  __bf16 v8bf;
typedef __attribute__((ext_vector_type(8)))  float  v8f;
typedef __attribute__((ext_vector_type(4)))  unsigned int u32x4;
typedef __attribute__((ext_vector_type(8)))  int  i32x8;
typedef __attribute__((ext_vector_type(4)))  int  i32x4;

#define D_MODEL 1024
#define N_HEADS 16
#define D_HEAD  64
#define BATCH   2
#define SEQ     2048
#define MTOT    (BATCH * SEQ)          // 4096
#define LOG2_DECAY (-0.014499569695115089f)  // log2(0.99)

#if __has_builtin(__builtin_amdgcn_tensor_load_to_lds)
#define USE_TDM 1
#else
#define USE_TDM 0
#endif

// ---- WMMA fragment helpers (layouts per CDNA5 ISA sec 7.12.2) ----
// A (16x32 bf16, M on lanes, K packed in VGPRs):
//   lanes 0-15 : row M=lane,    K = 0..7 and 16..23
//   lanes 16-31: row M=lane-16, K = 8..15 and 24..31
// B (32x16 bf16) mirrors A with N on lanes -> identical addressing from a
// row-major [N,K] buffer.
__device__ __forceinline__ v16bf load_frag(const __bf16* base, int ld) {
    const int lane = threadIdx.x & 31;
    const int row  = lane & 15;
    const int ksel = (lane >> 4) & 1;
    const __bf16* p = base + row * ld + ksel * 8;
    v8bf lo = *reinterpret_cast<const v8bf*>(p);
    v8bf hi = *reinterpret_cast<const v8bf*>(p + 16);
    v16bf f;
#pragma unroll
    for (int i = 0; i < 8; ++i) { f[i] = lo[i]; f[8 + i] = hi[i]; }
    return f;
}

__device__ __forceinline__ v8f zero_v8f() {
    v8f z;
#pragma unroll
    for (int i = 0; i < 8; ++i) z[i] = 0.0f;
    return z;
}

__device__ __forceinline__ v8f wmma_bf16(v16bf a, v16bf b, v8f c) {
    return __builtin_amdgcn_wmma_f32_16x16x32_bf16(false, a, false, b,
                                                   (short)0, c, false, false);
}

__device__ __forceinline__ unsigned lds_byte_off(const void* p) {
    // LDS flat address keeps the wave-relative byte offset in bits [31:0]
    return (unsigned)(unsigned long long)(uintptr_t)p;
}

#if USE_TDM
// Issue one TDM 2-D tile load (rows x kElems bf16, row-major both sides).
// D# layout per CDNA5 ISA (groups 2,3 zero: 2D).
// This toolchain exposes the 6-arg builtin:
//   (u32x4 g0, i32x8 g1, i32x4 g2, i32x4 g3, i32x8 gx, i32 cpol)
__device__ __forceinline__ void tdm_load_2d(unsigned lds_off, const void* gptr,
                                            unsigned kElems, unsigned rows,
                                            unsigned rowStrideElems) {
    const unsigned long long ga = (unsigned long long)(uintptr_t)gptr;
    u32x4 g0;
    g0[0] = 1u;                                        // count=1, user mode
    g0[1] = lds_off;                                   // lds_addr (bytes)
    g0[2] = (unsigned)(ga & 0xffffffffu);              // global_addr[31:0]
    g0[3] = (unsigned)((ga >> 32) & 0x1ffffffu)        // global_addr[56:32]
          | (2u << 30);                                // type = 2 (image)
    const unsigned td0 = 0x40000000u, td1 = 0x40000000u;  // huge dims: no OOB
    const unsigned long long s0 = (unsigned long long)rowStrideElems;
    i32x8 g1;
    g1[0] = (int)(1u << 16);                           // data_size=1 (2 bytes)
    g1[1] = (int)((td0 & 0xffffu) << 16);              // tensor_dim0[15:0]
    g1[2] = (int)((td0 >> 16) | ((td1 & 0xffffu) << 16));
    g1[3] = (int)((td1 >> 16) | (kElems << 16));       // tile_dim0
    g1[4] = (int)(rows & 0xffffu);                     // tile_dim1 (tile_dim2=0)
    g1[5] = (int)(s0 & 0xffffffffu);                   // tensor_dim0_stride lo
    g1[6] = (int)((s0 >> 32) & 0xffffu);               // stride hi | stride1 lo
    g1[7] = 0;
    i32x4 g2 = {0, 0, 0, 0};
    i32x4 g3 = {0, 0, 0, 0};
    i32x8 gx = {0, 0, 0, 0, 0, 0, 0, 0};
    __builtin_amdgcn_tensor_load_to_lds(g0, g1, g2, g3, gx, 0);
}
#endif

// ---------------------------------------------------------------------------
__global__ void k_convert_bf16(const float* __restrict__ src,
                               __bf16* __restrict__ dst, int n) {
    int i = blockIdx.x * blockDim.x + threadIdx.x;
    if (i < n) dst[i] = (__bf16)src[i];
}

// W [K,N] f32 row-major -> Wt [N,K] bf16 row-major (= W^T)
__global__ void k_transpose_bf16(const float* __restrict__ W,
                                 __bf16* __restrict__ Wt) {
    __shared__ float tile[32][33];
    const int bx = blockIdx.x * 32, by = blockIdx.y * 32;
    const int tx = threadIdx.x, ty = threadIdx.y;
#pragma unroll
    for (int i = 0; i < 32; i += 8)
        tile[ty + i][tx] = W[(by + ty + i) * D_MODEL + bx + tx];
    __syncthreads();
#pragma unroll
    for (int i = 0; i < 32; i += 8)
        Wt[(bx + ty + i) * D_MODEL + by + tx] = (__bf16)tile[tx][ty + i];
}

// ---------------------------------------------------------------------------
// GEMM: C[M=4096,N=1024] = A[M,K=1024](bf16) @ Bt[N,K](bf16)^T + bias
// Block 256 thr (8 waves), block tile 64M x 256N; wave tile 32M x 64N (8 WMMA
// per 32-K step). A/B tiles staged global->LDS by TDM (wave 0), double
// buffered, consumers synchronized via s_wait_tensorcnt + barrier.
// mode 0: bf16 -> [B,H,S,DH] (Q,K)   mode 1: f32 sigmoid (gate)
// mode 2: bf16 -> [B,H,DH,S] (V)     mode 3: f32 (final out)
// grid (4, 64)
// ---------------------------------------------------------------------------
__global__ void k_gemm(const __bf16* __restrict__ A,
                       const __bf16* __restrict__ Bt,
                       const float* __restrict__ bias,
                       void* __restrict__ dst, int mode) {
    __shared__ __attribute__((aligned(16))) __bf16 tA[2][64 * 32];    // 4 KB x2
    __shared__ __attribute__((aligned(16))) __bf16 tB[2][256 * 32];   // 16 KB x2

    const int wave = threadIdx.x >> 5;
    const int lane = threadIdx.x & 31;
    const int wm = wave >> 2;          // 0..1  (M sub-block of 32 rows)
    const int wn = wave & 3;           // 0..3  (N sub-block of 64 cols)
    const int n0 = blockIdx.x * 256;
    const int m0 = blockIdx.y * 64;

    v8f acc[2][4];
#pragma unroll
    for (int i = 0; i < 2; ++i)
#pragma unroll
        for (int t = 0; t < 4; ++t) acc[i][t] = zero_v8f();

#if USE_TDM
    if (wave == 0) {
        tdm_load_2d(lds_byte_off(&tA[0][0]), A + (size_t)m0 * D_MODEL, 32, 64, D_MODEL);
        tdm_load_2d(lds_byte_off(&tB[0][0]), Bt + (size_t)n0 * D_MODEL, 32, 256, D_MODEL);
    }
    int cur = 0;
    for (int k0 = 0; k0 < D_MODEL; k0 += 32) {
        const int nxt = cur ^ 1;
        const bool more = (k0 + 32) < D_MODEL;
        if (wave == 0) {
            if (more) {
                tdm_load_2d(lds_byte_off(&tA[nxt][0]),
                            A + (size_t)m0 * D_MODEL + k0 + 32, 32, 64, D_MODEL);
                tdm_load_2d(lds_byte_off(&tB[nxt][0]),
                            Bt + (size_t)n0 * D_MODEL + k0 + 32, 32, 256, D_MODEL);
                __builtin_amdgcn_s_wait_tensorcnt(2);  // current buffer done
            } else {
                __builtin_amdgcn_s_wait_tensorcnt(0);
            }
        }
        __syncthreads();   // publish buffer `cur` to all waves
        v16bf a0 = load_frag(&tA[cur][(wm * 32 + 0) * 32], 32);
        v16bf a1 = load_frag(&tA[cur][(wm * 32 + 16) * 32], 32);
#pragma unroll
        for (int t = 0; t < 4; ++t) {
            v16bf b = load_frag(&tB[cur][(wn * 64 + 16 * t) * 32], 32);
            acc[0][t] = wmma_bf16(a0, b, acc[0][t]);
            acc[1][t] = wmma_bf16(a1, b, acc[1][t]);
        }
        __syncthreads();   // all done reading `cur` before it is overwritten
        cur = nxt;
    }
#else
    // Fallback: cooperative global->LDS staging with plain loads.
    for (int k0 = 0; k0 < D_MODEL; k0 += 32) {
        for (int r = threadIdx.x; r < 64; r += 256) {
            v8bf* d = (v8bf*)&tA[0][r * 32];
            const v8bf* s = (const v8bf*)(A + (size_t)(m0 + r) * D_MODEL + k0);
            d[0] = s[0]; d[1] = s[1]; d[2] = s[2]; d[3] = s[3];
        }
        for (int r = threadIdx.x; r < 256; r += 256) {
            v8bf* d = (v8bf*)&tB[0][r * 32];
            const v8bf* s = (const v8bf*)(Bt + (size_t)(n0 + r) * D_MODEL + k0);
            d[0] = s[0]; d[1] = s[1]; d[2] = s[2]; d[3] = s[3];
        }
        __syncthreads();
        v16bf a0 = load_frag(&tA[0][(wm * 32 + 0) * 32], 32);
        v16bf a1 = load_frag(&tA[0][(wm * 32 + 16) * 32], 32);
#pragma unroll
        for (int t = 0; t < 4; ++t) {
            v16bf b = load_frag(&tB[0][(wn * 64 + 16 * t) * 32], 32);
            acc[0][t] = wmma_bf16(a0, b, acc[0][t]);
            acc[1][t] = wmma_bf16(a1, b, acc[1][t]);
        }
        __syncthreads();
    }
#endif

    // epilogue; C layout: elem r of lane -> (M = r + 8*(lane>=16), N = lane&15)
    const int nin = lane & 15;
    const int mb  = (lane >> 4) * 8;
#pragma unroll
    for (int i = 0; i < 2; ++i) {
#pragma unroll
        for (int t = 0; t < 4; ++t) {
            const int nc0 = n0 + wn * 64 + 16 * t;
#pragma unroll
            for (int r = 0; r < 8; ++r) {
                const int m = m0 + wm * 32 + 16 * i + mb + r;
                const int n = nc0 + nin;
                float v = acc[i][t][r] + bias[n];
                if (mode == 0) {
                    const int b = m >> 11, s = m & 2047, h = n >> 6, d = n & 63;
                    ((__bf16*)dst)[(((b * N_HEADS + h) * SEQ) + s) * D_HEAD + d] = (__bf16)v;
                } else if (mode == 1) {
                    ((float*)dst)[(size_t)m * D_MODEL + n] = 1.0f / (1.0f + __expf(-v));
                } else if (mode == 2) {
                    const int b = m >> 11, s = m & 2047, h = n >> 6, d = n & 63;
                    ((__bf16*)dst)[(((b * N_HEADS + h) * D_HEAD) + d) * SEQ + s] = (__bf16)v;
                } else {
                    ((float*)dst)[(size_t)m * D_MODEL + n] = v;
                }
            }
        }
    }
}

// ---------------------------------------------------------------------------
// Retention: per (b,h, 16-row tile) wave: out[16x64] =
//   sum_j (Q K^T / 8 * 0.99^(i-j) [j<=i]) V   (flash-style, 32-wide j tiles)
// Q,K bf16 [B,H,S,64]; Vt bf16 [B,H,64,S]; R f32 [B,S,1024]. grid 1024 x 128.
// ---------------------------------------------------------------------------
__global__ void k_retention(const __bf16* __restrict__ Q,
                            const __bf16* __restrict__ K,
                            const __bf16* __restrict__ Vt,
                            float* __restrict__ R) {
    __shared__ float dp[SEQ];   // 0.99^i * 0.125
    __shared__ float dn[SEQ];   // 0.99^(-j)
    __shared__ __attribute__((aligned(16))) __bf16 scr[4][16 * 32];

    for (int i = threadIdx.x; i < SEQ; i += blockDim.x) {
        dp[i] = exp2f((float)i * LOG2_DECAY) * 0.125f;
        dn[i] = exp2f(-(float)i * LOG2_DECAY);
    }
    __syncthreads();

    const int wave = threadIdx.x >> 5;
    const int lane = threadIdx.x & 31;
    const int item = blockIdx.x * 4 + wave;   // 0..4095
    const int i0 = (item & 127) * 16;
    const int bh = item >> 7;                 // 0..31
    const __bf16* Qh = Q + (size_t)bh * SEQ * D_HEAD;
    const __bf16* Kh = K + (size_t)bh * SEQ * D_HEAD;
    const __bf16* Vh = Vt + (size_t)bh * D_HEAD * SEQ;

    const v16bf aq0 = load_frag(Qh + i0 * D_HEAD + 0, D_HEAD);
    const v16bf aq1 = load_frag(Qh + i0 * D_HEAD + 32, D_HEAD);

    v8f acc[4];
#pragma unroll
    for (int d = 0; d < 4; ++d) acc[d] = zero_v8f();

    const int nin = lane & 15;
    const int mb  = (lane >> 4) * 8;
    float dprow[8];
#pragma unroll
    for (int r = 0; r < 8; ++r) dprow[r] = dp[i0 + mb + r];

    __bf16* sw = &scr[wave][0];
    const int tmax = i0 >> 5;
    for (int tt = 0; tt <= tmax; ++tt) {
        const int j0 = tt * 32;
        __builtin_prefetch(Kh + (j0 + 32) * D_HEAD, 0, 3);
        __builtin_prefetch(Vh + j0 + 32, 0, 3);
#pragma unroll
        for (int half = 0; half < 2; ++half) {
            const int jb = j0 + 16 * half;
            v16bf bk0 = load_frag(Kh + jb * D_HEAD + 0, D_HEAD);
            v16bf bk1 = load_frag(Kh + jb * D_HEAD + 32, D_HEAD);
            v8f s = zero_v8f();
            s = wmma_bf16(aq0, bk0, s);
            s = wmma_bf16(aq1, bk1, s);
            const int j = jb + nin;
            const float dnj = dn[j];
#pragma unroll
            for (int r = 0; r < 8; ++r) {
                const int i = i0 + mb + r;
                const float w = (j <= i) ? dprow[r] * dnj : 0.0f;
                s[r] *= w;
            }
#pragma unroll
            for (int r = 0; r < 8; ++r)
                sw[(mb + r) * 32 + 16 * half + nin] = (__bf16)s[r];
        }
        v16bf aS = load_frag(sw, 32);   // same-wave DS ops are in-order
#pragma unroll
        for (int d = 0; d < 4; ++d) {
            v16bf bv = load_frag(Vh + (d * 16) * SEQ + j0, SEQ);
            acc[d] = wmma_bf16(aS, bv, acc[d]);
        }
    }

    const int b = bh >> 4, h = bh & 15;
#pragma unroll
    for (int d = 0; d < 4; ++d)
#pragma unroll
        for (int r = 0; r < 8; ++r) {
            const int i = i0 + mb + r;
            R[((size_t)(b * SEQ + i)) * D_MODEL + h * D_HEAD + d * 16 + nin] = acc[d][r];
        }
}

// ---------------------------------------------------------------------------
__global__ void k_ln_gate(const float* __restrict__ R,
                          const float* __restrict__ Gsig,
                          const float* __restrict__ gamma,
                          const float* __restrict__ beta,
                          __bf16* __restrict__ Y) {
    const int row = blockIdx.x;
    const float* r = R + (size_t)row * D_MODEL;
    float v[4];
    float sum = 0.0f, sq = 0.0f;
#pragma unroll
    for (int t = 0; t < 4; ++t) {
        v[t] = r[threadIdx.x + t * 256];
        sum += v[t];
        sq  += v[t] * v[t];
    }
#pragma unroll
    for (int off = 16; off > 0; off >>= 1) {
        sum += __shfl_xor(sum, off, 32);
        sq  += __shfl_xor(sq, off, 32);
    }
    __shared__ float rs[8], rq[8];
    const int wave = threadIdx.x >> 5;
    if ((threadIdx.x & 31) == 0) { rs[wave] = sum; rq[wave] = sq; }
    __syncthreads();
    float ts = 0.0f, tq = 0.0f;
#pragma unroll
    for (int w = 0; w < 8; ++w) { ts += rs[w]; tq += rq[w]; }
    const float mu  = ts * (1.0f / D_MODEL);
    const float var = tq * (1.0f / D_MODEL) - mu * mu;
    const float inv = rsqrtf(var + 1e-5f);
#pragma unroll
    for (int t = 0; t < 4; ++t) {
        const int c = threadIdx.x + t * 256;
        const float y = (v[t] - mu) * inv * gamma[c] + beta[c];
        Y[(size_t)row * D_MODEL + c] = (__bf16)(y * Gsig[(size_t)row * D_MODEL + c]);
    }
}

// ---------------------------------------------------------------------------
extern "C" void kernel_launch(void* const* d_in, const int* in_sizes, int n_in,
                              void* d_out, int out_size, void* d_ws, size_t ws_size,
                              hipStream_t stream) {
    (void)in_sizes; (void)n_in; (void)out_size; (void)ws_size;
    const float* x     = (const float*)d_in[0];
    const float* Wq    = (const float*)d_in[1];
    const float* bq    = (const float*)d_in[2];
    const float* Wk    = (const float*)d_in[3];
    const float* bk    = (const float*)d_in[4];
    const float* Wv    = (const float*)d_in[5];
    const float* bv    = (const float*)d_in[6];
    const float* Wg    = (const float*)d_in[7];
    const float* bg    = (const float*)d_in[8];
    const float* Wo    = (const float*)d_in[9];
    const float* bo    = (const float*)d_in[10];
    const float* gamma = (const float*)d_in[11];
    const float* beta  = (const float*)d_in[12];

    char* w = (char*)d_ws;
    const size_t SZ_XB = (size_t)MTOT * D_MODEL * 2;      // 8 MB
    const size_t SZ_WT = (size_t)D_MODEL * D_MODEL * 2;   // 2 MB
    __bf16* xb  = (__bf16*)w;             w += SZ_XB;
    __bf16* WqT = (__bf16*)w;             w += SZ_WT;
    __bf16* WkT = (__bf16*)w;             w += SZ_WT;
    __bf16* WvT = (__bf16*)w;             w += SZ_WT;
    __bf16* WgT = (__bf16*)w;             w += SZ_WT;
    __bf16* WoT = (__bf16*)w;             w += SZ_WT;
    __bf16* Qb  = (__bf16*)w;             w += SZ_XB;     // [B,H,S,64] bf16
    __bf16* Kb  = (__bf16*)w;             w += SZ_XB;
    __bf16* Vtb = (__bf16*)w;             w += SZ_XB;     // [B,H,64,S] bf16
    float*  Gsg = (float*)w;              w += (size_t)MTOT * D_MODEL * 4;
    float*  R   = (float*)w;              w += (size_t)MTOT * D_MODEL * 4;
    __bf16* Yb  = (__bf16*)w;             w += SZ_XB;

    {
        const int n = MTOT * D_MODEL;
        k_convert_bf16<<<(n + 255) / 256, 256, 0, stream>>>(x, xb, n);
    }
    {
        dim3 g(32, 32), b(32, 8);
        k_transpose_bf16<<<g, b, 0, stream>>>(Wq, WqT);
        k_transpose_bf16<<<g, b, 0, stream>>>(Wk, WkT);
        k_transpose_bf16<<<g, b, 0, stream>>>(Wv, WvT);
        k_transpose_bf16<<<g, b, 0, stream>>>(Wg, WgT);
        k_transpose_bf16<<<g, b, 0, stream>>>(Wo, WoT);
    }
    {
        dim3 g(D_MODEL / 256, MTOT / 64);   // (4, 64)
        k_gemm<<<g, 256, 0, stream>>>(xb, WqT, bq, (void*)Qb, 0);
        k_gemm<<<g, 256, 0, stream>>>(xb, WkT, bk, (void*)Kb, 0);
        k_gemm<<<g, 256, 0, stream>>>(xb, WvT, bv, (void*)Vtb, 2);
        k_gemm<<<g, 256, 0, stream>>>(xb, WgT, bg, (void*)Gsg, 1);
    }
    k_retention<<<(BATCH * N_HEADS * (SEQ / 16)) / 4, 128, 0, stream>>>(Qb, Kb, Vtb, R);
    k_ln_gate<<<MTOT, 256, 0, stream>>>(R, Gsg, gamma, beta, Yb);
    {
        dim3 g(D_MODEL / 256, MTOT / 64);
        k_gemm<<<g, 256, 0, stream>>>(Yb, WoT, bo, d_out, 3);
    }
}
